// TransformerBlock_67293547594346
// MI455X (gfx1250) — matrix operand order repaired and checked by reading, not compiled, not run on hardware
//
#include <hip/hip_runtime.h>

// ---------------------------------------------------------------------------
// MI455X (gfx1250, wave32) transformer block.
// Compute-bound (~0.62 TFLOP vs ~0.3 GB -> ~2000 FLOP/byte): all GEMMs +
// attention run on v_wmma_f32_16x16x32_bf16 (f32 accum). bf16 tiles are
// staged into LDS with GLOBAL_LOAD_ASYNC_TO_LDS_B128 (ASYNCcnt) and double
// buffered so WMMA overlaps the async DMA; fp32 weight tiles are converted
// to bf16 through VGPRs after the WMMAs are issued.
// Workspace layout (bytes):
//   [0,16M)     h / h2        bf16  B*S*D
//   [16M,32M)   q             bf16
//   [32M,48M)   k             bf16
//   [48M,64M)   v             bf16
//   [64M,80M)   attn out      bf16
//   [80M,112M)  x1 = x + o@WoT f32
//   [112M,176M) u = h2@W1T    bf16  B*S*DFF (silu*gate in place)
//   [176M,240M) g3 = h2@W3T   bf16
// ---------------------------------------------------------------------------

typedef __attribute__((ext_vector_type(16))) __bf16 v16bf;
typedef __attribute__((ext_vector_type(8)))  float  v8f;

union B16Frag { v16bf v; uint4 q[2]; };

__device__ __forceinline__ v8f wmma_bf16(const B16Frag& a, const B16Frag& b, v8f c) {
  return __builtin_amdgcn_wmma_f32_16x16x32_bf16(false, a.v, false, b.v, (short)0, c,
                                                 false, false);
}

__device__ __forceinline__ v8f zero_v8f() {
  v8f z;
#pragma unroll
  for (int i = 0; i < 8; ++i) z[i] = 0.0f;
  return z;
}

// Async memory->LDS copy of 16 bytes per lane (no VGPR data path, ASYNCcnt).
// INST_OFFSET is applied to BOTH the global and the LDS address (ISA 15.18.3),
// so one base covers consecutive chunks.
__device__ __forceinline__ void async_b128(const void* gptr, void* lptr) {
  asm volatile("global_load_async_to_lds_b128 %0, %1, off"
               :: "v"((uint32_t)(uintptr_t)lptr), "v"(gptr) : "memory");
}
__device__ __forceinline__ void async_b128_off(const void* gptr, void* lptr, int off16) {
  if (off16 == 0)
    asm volatile("global_load_async_to_lds_b128 %0, %1, off"
                 :: "v"((uint32_t)(uintptr_t)lptr), "v"(gptr) : "memory");
  else if (off16 == 1)
    asm volatile("global_load_async_to_lds_b128 %0, %1, off offset:16"
                 :: "v"((uint32_t)(uintptr_t)lptr), "v"(gptr) : "memory");
  else if (off16 == 2)
    asm volatile("global_load_async_to_lds_b128 %0, %1, off offset:32"
                 :: "v"((uint32_t)(uintptr_t)lptr), "v"(gptr) : "memory");
  else
    asm volatile("global_load_async_to_lds_b128 %0, %1, off offset:48"
                 :: "v"((uint32_t)(uintptr_t)lptr), "v"(gptr) : "memory");
}
__device__ __forceinline__ void wait_async0() {
  asm volatile("s_wait_asynccnt 0x0" ::: "memory");
}

// ---------------------------------------------------------------------------
// RMSNorm: one block per row, fp32 in, bf16 out.
// ---------------------------------------------------------------------------
__global__ __launch_bounds__(256) void tb_rmsnorm_bf16(const float* __restrict__ x,
                                                       const float* __restrict__ w,
                                                       __bf16* __restrict__ out, int D) {
  const int row = blockIdx.x;
  const float* xr = x + (size_t)row * D;
  __bf16* orow = out + (size_t)row * D;
  float ss = 0.0f;
  for (int i = threadIdx.x; i < D; i += 256) {
    float v = xr[i];
    ss += v * v;
  }
#pragma unroll
  for (int off = 16; off > 0; off >>= 1) ss += __shfl_xor(ss, off, 32);
  __shared__ float wsum[8];
  if ((threadIdx.x & 31) == 0) wsum[threadIdx.x >> 5] = ss;
  __syncthreads();
  float tot = 0.0f;
#pragma unroll
  for (int i = 0; i < 8; ++i) tot += wsum[i];
  const float scale = rsqrtf(tot / (float)D + 1e-5f);
  for (int i = threadIdx.x; i < D; i += 256) orow[i] = (__bf16)(xr[i] * scale * w[i]);
}

// ---------------------------------------------------------------------------
// Tiled WMMA GEMM:  C[M,N] = A[M,K](bf16) @ W[N,K](f32)^T   (+ resid, f32)
// Block tile 128x128, BK=32, 256 threads = 8 waves, each wave 64x32 (4x2 frags).
// Double-buffered LDS; A tile staged via async-to-LDS, W tile via VGPR cvt.
// ---------------------------------------------------------------------------
template <bool STORE_BF16>
__global__ __launch_bounds__(256) void tb_gemm_bf16w(const __bf16* __restrict__ A,
                                                     const float* __restrict__ W,
                                                     const float* __restrict__ resid,
                                                     float* __restrict__ outF,
                                                     __bf16* __restrict__ outB,
                                                     int M, int N, int K) {
  __shared__ __bf16 As[2][128][40];  // row stride 80 B (16-aligned), 10 KB/buf
  __shared__ __bf16 Bs[2][128][40];

  const int t = threadIdx.x;
  const int m0 = blockIdx.y * 128;
  const int n0 = blockIdx.x * 128;
  const int wid = t >> 5, lane = t & 31;
  const int hsel = lane >> 4, lm = lane & 15;
  const int wm = wid & 1;   // 2 row groups of 64
  const int wn = wid >> 1;  // 4 col groups of 32

  v8f acc[4][2];
#pragma unroll
  for (int i = 0; i < 4; ++i)
#pragma unroll
    for (int j = 0; j < 2; ++j) acc[i][j] = zero_v8f();

  const int lrow = t >> 1;        // 0..127
  const int lcol = (t & 1) * 16;  // 0 or 16
  const __bf16* agBase = A + (size_t)(m0 + lrow) * K + lcol;
  const float*  wgBase = W + (size_t)(n0 + lrow) * K + lcol;

  // ---- prologue: stage tile k0=0 into buffer 0 ----
  {
    async_b128_off(agBase, &As[0][lrow][lcol], 0);
    async_b128_off(agBase, &As[0][lrow][lcol], 1);
    float wf[16];
    ((float4*)wf)[0] = ((const float4*)wgBase)[0];
    ((float4*)wf)[1] = ((const float4*)wgBase)[1];
    ((float4*)wf)[2] = ((const float4*)wgBase)[2];
    ((float4*)wf)[3] = ((const float4*)wgBase)[3];
    union { __bf16 b[16]; uint4 q[2]; } cvt;
#pragma unroll
    for (int e = 0; e < 16; ++e) cvt.b[e] = (__bf16)wf[e];
    *(uint4*)&Bs[0][lrow][lcol] = cvt.q[0];
    *(uint4*)&Bs[0][lrow][lcol + 8] = cvt.q[1];
    wait_async0();
    __syncthreads();
  }

  int cb = 0;
  for (int k0 = 0; k0 < K; k0 += 32) {
    const int nb = cb ^ 1;
    const bool has_next = (k0 + 32) < K;

    // ---- issue next-tile staging first (latency hiding) ----
    float wf[16];
    if (has_next) {
      const float* wg = wgBase + (k0 + 32);
      ((float4*)wf)[0] = ((const float4*)wg)[0];
      ((float4*)wf)[1] = ((const float4*)wg)[1];
      ((float4*)wf)[2] = ((const float4*)wg)[2];
      ((float4*)wf)[3] = ((const float4*)wg)[3];
      const __bf16* ag = agBase + (k0 + 32);
      async_b128_off(ag, &As[nb][lrow][lcol], 0);
      async_b128_off(ag, &As[nb][lrow][lcol], 1);
    }

    // ---- compute on current buffer: 4 A frags x 2 B frags -> 8 WMMAs ----
    B16Frag af[4], bf[2];
#pragma unroll
    for (int i = 0; i < 4; ++i) {
      const __bf16* ar = &As[cb][wm * 64 + i * 16 + lm][0];
      af[i].q[0] = *(const uint4*)(ar + 16 * hsel);      // k = 16h..16h+7
      af[i].q[1] = *(const uint4*)(ar + 16 * hsel + 8);  // k = 16h+8..16h+15
    }
#pragma unroll
    for (int j = 0; j < 2; ++j) {
      const __bf16* br = &Bs[cb][wn * 32 + j * 16 + lm][16 * hsel];  // k = 16h..16h+15
      bf[j].q[0] = *(const uint4*)br;
      bf[j].q[1] = *(const uint4*)(br + 8);
    }
#pragma unroll
    for (int i = 0; i < 4; ++i)
#pragma unroll
      for (int j = 0; j < 2; ++j) acc[i][j] = wmma_bf16(af[i], bf[j], acc[i][j]);

    // ---- finish next-tile W staging (cvt after WMMAs are issued) ----
    if (has_next) {
      union { __bf16 b[16]; uint4 q[2]; } cvt;
#pragma unroll
      for (int e = 0; e < 16; ++e) cvt.b[e] = (__bf16)wf[e];
      *(uint4*)&Bs[nb][lrow][lcol] = cvt.q[0];
      *(uint4*)&Bs[nb][lrow][lcol + 8] = cvt.q[1];
      wait_async0();
      __syncthreads();
    }
    cb = nb;
  }

  // --- epilogue: C layout: n = lane&15, m = r + 8*(lane>>4) ---
#pragma unroll
  for (int i = 0; i < 4; ++i) {
#pragma unroll
    for (int j = 0; j < 2; ++j) {
      const int gn = n0 + wn * 32 + j * 16 + lm;
      const int gmb = m0 + wm * 64 + i * 16 + 8 * hsel;
#pragma unroll
      for (int r = 0; r < 8; ++r) {
        const size_t idx = (size_t)(gmb + r) * N + gn;
        float vacc = acc[i][j][r];
        if (STORE_BF16) {
          outB[idx] = (__bf16)vacc;
        } else {
          if (resid) vacc += resid[idx];
          outF[idx] = vacc;
        }
      }
    }
  }
}

// ---------------------------------------------------------------------------
// Interleaved RoPE, in place on bf16 q and k.  Layout [B,S,H,DK] flat = [B*S, D].
// ---------------------------------------------------------------------------
__global__ __launch_bounds__(256) void tb_rope_inplace(__bf16* __restrict__ q,
                                                       __bf16* __restrict__ k,
                                                       const int* __restrict__ pos,
                                                       size_t npairs) {
  constexpr int S = 2048, D = 2048, DK = 128;
  size_t idx = (size_t)blockIdx.x * 256 + threadIdx.x;
  if (idx >= npairs) return;
  const int halfD = D >> 1;
  const size_t row = idx / halfD;
  const int p = (int)(idx % halfD);
  const int hd = p / (DK / 2);
  const int i = p % (DK / 2);
  const int s = (int)(row % S);
  const float inv = __expf((-2.0f * (float)i / (float)DK) * logf(10000.0f));
  const float ang = (float)pos[s] * inv;
  float sn, cs;
  sincosf(ang, &sn, &cs);
  const size_t base = row * (size_t)D + hd * DK + 2 * i;
  {
    float x1 = (float)q[base], x2 = (float)q[base + 1];
    q[base] = (__bf16)(x1 * cs - x2 * sn);
    q[base + 1] = (__bf16)(x1 * sn + x2 * cs);
  }
  {
    float x1 = (float)k[base], x2 = (float)k[base + 1];
    k[base] = (__bf16)(x1 * cs - x2 * sn);
    k[base + 1] = (__bf16)(x1 * sn + x2 * cs);
  }
}

// ---------------------------------------------------------------------------
// Flash attention, causal.  Grid (S/64, B*H), 256 threads = 8 waves.
// Q/K tiles staged with async-to-LDS; V transposed through VGPRs.
// ---------------------------------------------------------------------------
__global__ __launch_bounds__(256) void tb_flash_attn(const __bf16* __restrict__ q,
                                                     const __bf16* __restrict__ k,
                                                     const __bf16* __restrict__ v,
                                                     __bf16* __restrict__ o) {
  constexpr int S = 2048, D = 2048, H = 16, DK = 128;
  __shared__ __bf16 Qs[64][136];   // 272 B row stride (16-aligned)
  __shared__ __bf16 Ks[64][136];
  __shared__ __bf16 Vts[128][72];  // V^T: [d][n]
  __shared__ float Ss[64][68];
  __shared__ __bf16 Ps[64][72];
  __shared__ float rowScale[64];
  __shared__ float rowInv[64];

  const int qt = blockIdx.x;
  const int bh = blockIdx.y;
  const int b = bh >> 4;  // H == 16
  const int hd = bh & 15;
  const int t = threadIdx.x;
  const int wid = t >> 5, lane = t & 31;
  const int hsel = lane >> 4, lm = lane & 15;
  const int rs = (wid & 3) * 16;  // row strip in [0,64)
  const int ch = wid >> 2;        // 0/1: score col half (32) / output d half (64)

  const size_t baseQ = ((size_t)(b * S + qt * 64)) * D + (size_t)hd * DK;
  const __bf16* Qg = q + baseQ;
  __bf16* Og = o + baseQ;

  const int lr = t >> 2;         // 0..63
  const int lc = (t & 3) * 32;   // 0/32/64/96

  {  // async-load Q tile (64 x 128), 64 B per lane
    const __bf16* src = Qg + (size_t)lr * D + lc;
    void* dst = &Qs[lr][lc];
    async_b128_off(src, dst, 0);
    async_b128_off(src, dst, 1);
    async_b128_off(src, dst, 2);
    async_b128_off(src, dst, 3);
  }

  float m_i = -1e30f, l_i = 0.0f;  // live in threads 0..63
  v8f oacc[4];
#pragma unroll
  for (int tt = 0; tt < 4; ++tt) oacc[tt] = zero_v8f();

  for (int kt = 0; kt <= qt; ++kt) {
    const size_t baseK = ((size_t)(b * S + kt * 64)) * D + (size_t)hd * DK;
    {  // async K tile; V^T through VGPRs (transpose)
      const __bf16* ksrc = k + baseK + (size_t)lr * D + lc;
      void* kdst = &Ks[lr][lc];
      async_b128_off(ksrc, kdst, 0);
      async_b128_off(ksrc, kdst, 1);
      async_b128_off(ksrc, kdst, 2);
      async_b128_off(ksrc, kdst, 3);
      const __bf16* vsrc = v + baseK + (size_t)lr * D + lc;
      union { uint4 q4[4]; __bf16 b16[32]; } tmp;
#pragma unroll
      for (int u = 0; u < 4; ++u) tmp.q4[u] = ((const uint4*)vsrc)[u];
#pragma unroll
      for (int dd = 0; dd < 32; ++dd) Vts[lc + dd][lr] = tmp.b16[dd];
    }
    wait_async0();
    __syncthreads();

    // scores S = Q K^T  (per wave: 16 rows x 32 cols)
    v8f sacc[2];
    sacc[0] = zero_v8f();
    sacc[1] = zero_v8f();
#pragma unroll
    for (int k0 = 0; k0 < DK; k0 += 32) {
      B16Frag a;
      const __bf16* ar = &Qs[rs + lm][k0];
      a.q[0] = *(const uint4*)(ar + 16 * hsel);
      a.q[1] = *(const uint4*)(ar + 16 * hsel + 8);
#pragma unroll
      for (int j = 0; j < 2; ++j) {
        B16Frag bb;
        const __bf16* kr = &Ks[ch * 32 + j * 16 + lm][k0 + 16 * hsel];
        bb.q[0] = *(const uint4*)kr;
        bb.q[1] = *(const uint4*)(kr + 8);
        sacc[j] = wmma_bf16(a, bb, sacc[j]);
      }
    }
    const float sm = 0.08838834764831845f;  // 1/sqrt(128)
#pragma unroll
    for (int j = 0; j < 2; ++j)
#pragma unroll
      for (int r = 0; r < 8; ++r)
        Ss[rs + r + 8 * hsel][ch * 32 + j * 16 + lm] = sacc[j][r] * sm;
    __syncthreads();

    // online softmax (one thread per row)
    if (t < 64) {
      const int r = t;
      const int nvalid = (kt == qt) ? (r + 1) : 64;
      float mt = -1e30f;
      for (int n = 0; n < nvalid; ++n) mt = fmaxf(mt, Ss[r][n]);
      const float mnew = fmaxf(m_i, mt);
      const float alpha = __expf(m_i - mnew);
      float psum = 0.0f;
      for (int n = 0; n < 64; ++n) {
        const float pv = (n < nvalid) ? __expf(Ss[r][n] - mnew) : 0.0f;
        Ps[r][n] = (__bf16)pv;
        psum += pv;
      }
      l_i = l_i * alpha + psum;
      m_i = mnew;
      rowScale[r] = alpha;
    }
    __syncthreads();

    // rescale running O, then O += P @ V  (per wave: 16 rows x 64 d-cols)
#pragma unroll
    for (int tt = 0; tt < 4; ++tt)
#pragma unroll
      for (int r = 0; r < 8; ++r) oacc[tt][r] *= rowScale[rs + r + 8 * hsel];
#pragma unroll
    for (int k0 = 0; k0 < 64; k0 += 32) {
      B16Frag a;
      const __bf16* pr = &Ps[rs + lm][k0];
      a.q[0] = *(const uint4*)(pr + 16 * hsel);
      a.q[1] = *(const uint4*)(pr + 16 * hsel + 8);
#pragma unroll
      for (int tt = 0; tt < 4; ++tt) {
        B16Frag bb;
        const __bf16* vr = &Vts[ch * 64 + tt * 16 + lm][k0 + 16 * hsel];
        bb.q[0] = *(const uint4*)vr;
        bb.q[1] = *(const uint4*)(vr + 8);
        oacc[tt] = wmma_bf16(a, bb, oacc[tt]);
      }
    }
    __syncthreads();
  }

  if (t < 64) rowInv[t] = (l_i > 0.0f) ? (1.0f / l_i) : 0.0f;
  __syncthreads();
#pragma unroll
  for (int tt = 0; tt < 4; ++tt)
#pragma unroll
    for (int r = 0; r < 8; ++r) {
      const int row = rs + r + 8 * hsel;
      Og[(size_t)row * D + ch * 64 + tt * 16 + lm] = (__bf16)(oacc[tt][r] * rowInv[row]);
    }
}

// ---------------------------------------------------------------------------
// m = silu(u) * g, in place into u (bf16).
// ---------------------------------------------------------------------------
__global__ __launch_bounds__(256) void tb_silu_mul(__bf16* __restrict__ u,
                                                   const __bf16* __restrict__ g, size_t n) {
  const size_t i = (size_t)blockIdx.x * 256 + threadIdx.x;
  if (i >= n) return;
  const float a = (float)u[i];
  const float b = (float)g[i];
  u[i] = (__bf16)((a / (1.0f + __expf(-a))) * b);
}

// ---------------------------------------------------------------------------
extern "C" void kernel_launch(void* const* d_in, const int* in_sizes, int n_in,
                              void* d_out, int out_size, void* d_ws, size_t ws_size,
                              hipStream_t stream) {
  (void)in_sizes; (void)n_in; (void)out_size; (void)ws_size;
  constexpr int B = 2, S = 2048, D = 2048, H = 16, DFF = 8192;
  constexpr int BS = B * S;

  const float* x    = (const float*)d_in[0];
  const int*   pos  = (const int*)d_in[1];
  const float* ln1w = (const float*)d_in[2];
  const float* ln2w = (const float*)d_in[3];
  const float* wq   = (const float*)d_in[4];
  const float* wk   = (const float*)d_in[5];
  const float* wv   = (const float*)d_in[6];
  const float* wo   = (const float*)d_in[7];
  const float* w1   = (const float*)d_in[8];
  const float* w2   = (const float*)d_in[9];
  const float* w3   = (const float*)d_in[10];

  char* ws = (char*)d_ws;
  const size_t MB = 1ull << 20;
  __bf16* hbf = (__bf16*)(ws + 0);        // h / h2
  __bf16* qb  = (__bf16*)(ws + 16 * MB);
  __bf16* kb  = (__bf16*)(ws + 32 * MB);
  __bf16* vb  = (__bf16*)(ws + 48 * MB);
  __bf16* ob  = (__bf16*)(ws + 64 * MB);
  float*  x1  = (float*)(ws + 80 * MB);   // f32, 32 MB
  __bf16* ub  = (__bf16*)(ws + 112 * MB); // 64 MB
  __bf16* g3  = (__bf16*)(ws + 176 * MB); // 64 MB
  float* out = (float*)d_out;

  const dim3 blk(256);
  const dim3 gD(D / 128, BS / 128);    // (16, 32)
  const dim3 gF(DFF / 128, BS / 128);  // (64, 32)

  // h = rmsnorm(x) -> bf16
  tb_rmsnorm_bf16<<<BS, 256, 0, stream>>>(x, ln1w, hbf, D);
  // q/k/v = h @ W^T  (bf16 out)
  tb_gemm_bf16w<true><<<gD, blk, 0, stream>>>(hbf, wq, nullptr, nullptr, qb, BS, D, D);
  tb_gemm_bf16w<true><<<gD, blk, 0, stream>>>(hbf, wk, nullptr, nullptr, kb, BS, D, D);
  tb_gemm_bf16w<true><<<gD, blk, 0, stream>>>(hbf, wv, nullptr, nullptr, vb, BS, D, D);
  // rope(q), rope(k) in place
  const size_t npairs = (size_t)BS * D / 2;
  tb_rope_inplace<<<(unsigned)((npairs + 255) / 256), 256, 0, stream>>>(qb, kb, pos, npairs);
  // causal flash attention
  tb_flash_attn<<<dim3(S / 64, B * H), blk, 0, stream>>>(qb, kb, vb, ob);
  // x1 = x + attn @ Wo^T  (f32 out, fused residual)
  tb_gemm_bf16w<false><<<gD, blk, 0, stream>>>(ob, wo, x, x1, nullptr, BS, D, D);
  // h2 = rmsnorm(x1) -> bf16 (reuse h buffer)
  tb_rmsnorm_bf16<<<BS, 256, 0, stream>>>(x1, ln2w, hbf, D);
  // u = h2 @ W1^T, g3 = h2 @ W3^T
  tb_gemm_bf16w<true><<<gF, blk, 0, stream>>>(hbf, w1, nullptr, nullptr, ub, BS, DFF, D);
  tb_gemm_bf16w<true><<<gF, blk, 0, stream>>>(hbf, w3, nullptr, nullptr, g3, BS, DFF, D);
  // u = silu(u) * g3
  const size_t nffn = (size_t)BS * DFF;
  tb_silu_mul<<<(unsigned)((nffn + 255) / 256), 256, 0, stream>>>(ub, g3, nffn);
  // out = x1 + u @ W2^T  (f32 out, fused residual)
  tb_gemm_bf16w<false><<<gD, blk, 0, stream>>>(ub, w2, x1, out, nullptr, BS, D, DFF);
}
